// SimpleHierarchicalAttention_39376260169970
// MI455X (gfx1250) — matrix operand order repaired and checked by reading, not compiled, not run on hardware
//
#include <hip/hip_runtime.h>
#include <hip/hip_bf16.h>

#define BS 8
#define S1 64
#define S2 128
#define H  768
#define QS 64
#define NEGV (-10000.0f)

typedef __attribute__((ext_vector_type(16))) __bf16 v16bf;
typedef __attribute__((ext_vector_type(8)))  float  v8f;

union AFrag { v16bf v; uint4 q[2]; };
union BFrag { v16bf v; __bf16 b[16]; };

// ---- optional gfx1250 async global->LDS path, guarded so compile never fails
#if defined(__has_builtin)
#if __has_builtin(__builtin_amdgcn_global_load_async_to_lds_b32) && \
    __has_builtin(__builtin_amdgcn_s_wait_asynccnt)
#define USE_ASYNC_LDS 1
#endif
#endif
#ifndef USE_ASYNC_LDS
#define USE_ASYNC_LDS 0
#endif

#if USE_ASYNC_LDS
typedef __attribute__((address_space(3))) int lds_int;
typedef __attribute__((address_space(1))) int gbl_int;
__device__ __forceinline__ lds_int* to_lds(void* p) {
  // generic LDS pointer carries the LDS byte offset in its low 32 bits
  return (lds_int*)(unsigned)(uintptr_t)p;
}
__device__ __forceinline__ gbl_int* to_gbl(const void* p) {
  return (gbl_int*)(uintptr_t)p;
}
#endif

// ---------------------------------------------------------------- zero output
__global__ void zero_out_kernel(float* __restrict__ out, int n) {
  int i = blockIdx.x * blockDim.x + threadIdx.x;
  if (i < n) out[i] = 0.0f;
}

// ------------------------------------------------- prep: bf16(Q) + sent_prob
// grid = BS*QS blocks of 64 threads; block handles one (b,q) row.
__global__ void prep_kernel(const float* __restrict__ query,
                            const float* __restrict__ key_vec,
                            const int*   __restrict__ sentence_mask,
                            __bf16* __restrict__ qbf,
                            float* __restrict__ sentp) {
  int bq = blockIdx.x;          // b*QS + q
  int b  = bq / QS;
  int t  = threadIdx.x;         // 0..63 : sentence index s
  __shared__ float qrow[H];
  __shared__ float buf[S1];

  const float* qp = query + (size_t)bq * H;
  for (int i = t; i < H; i += 64) {
    float v = qp[i];
    qrow[i] = v;
    qbf[(size_t)bq * H + i] = (__bf16)v;    // hw RNE convert
  }
  __syncthreads();

  const float* kv = key_vec + (size_t)(b * S1 + t) * H;
  float dot = 0.0f;
  for (int i = 0; i < H; ++i) dot = fmaf(qrow[i], kv[i], dot);
  float sm = (1.0f - (float)sentence_mask[bq * S1 + t]) * NEGV;
  float sc = dot + sm;

  buf[t] = sc;
  __syncthreads();
  float mx = buf[0];
  for (int j = 1; j < S1; ++j) mx = fmaxf(mx, buf[j]);
  float e = __expf(sc - mx);
  __syncthreads();
  buf[t] = e;
  __syncthreads();
  float sum = 0.0f;
  for (int j = 0; j < S1; ++j) sum += buf[j];
  sentp[bq * S1 + t] = e / sum;
}

// ------------------------------------------------------------- fused attention
// grid = BS*S1 blocks (one per (b,s)), 256 threads = 8 wave32.
__global__ __launch_bounds__(256)
void attn_kernel(const __bf16* __restrict__ qbf,
                 const float* __restrict__ key,
                 const float* __restrict__ value,
                 const int*   __restrict__ word_mask,
                 const float* __restrict__ sentp,
                 float* __restrict__ out) {
  constexpr int LDS_S = S2 + 4;   // f32 score rows, conflict-spread
  constexpr int LDP   = S2 + 8;   // bf16 P rows, 16B-aligned stride (272B)
  __shared__ float  s_scores[QS][LDS_S];
  __shared__ __bf16 s_p[QS][LDP];
  __shared__ float  s_sp[QS];
  __shared__ int    s_wm[S2];

  int b    = blockIdx.x / S1;
  int s    = blockIdx.x % S1;
  int wave = threadIdx.x >> 5;
  int lane = threadIdx.x & 31;
  int ln   = lane & 15;
  int kh   = lane >> 4;          // 0 or 1 (lane half)

  const __bf16* qb = qbf + (size_t)(b * QS) * H;
  const float* Kb = key   + (size_t)(b * S1 + s) * S2 * H;
  const float* Vb = value + (size_t)(b * S1 + s) * S2 * H;
  const int* wmp = word_mask + (size_t)(b * S1 + s) * S2;

#if USE_ASYNC_LDS
  // Kick off side-loads asynchronously; they complete under the QK GEMM loop.
  if (kh == 0) {
    int c = 16 * wave + ln;
    __builtin_amdgcn_global_load_async_to_lds_b32(
        to_gbl(wmp + c), to_lds(&s_wm[c]), 0, 0);
    __builtin_amdgcn_global_load_async_to_lds_b32(
        to_gbl(wmp + 64 + c), to_lds(&s_wm[64 + c]), 0, 0);
  }
  if (threadIdx.x < QS) {
    __builtin_amdgcn_global_load_async_to_lds_b32(
        to_gbl(sentp + (b * QS + threadIdx.x) * S1 + s),
        to_lds(&s_sp[threadIdx.x]), 0, 0);
  }
#else
  if (threadIdx.x < QS)
    s_sp[threadIdx.x] = sentp[(b * QS + threadIdx.x) * S1 + s];
  if (threadIdx.x < S2)
    s_wm[threadIdx.x] = wmp[threadIdx.x];
#endif

  // ---- Phase 1: S = Q @ K^T ; wave owns s2 columns [16*wave, 16*wave+16)
  {
    int col = 16 * wave + ln;
    const float* krow = Kb + (size_t)col * H + 16 * kh;   // B lane: 16 contig h
    v8f acc0 = {}, acc1 = {}, acc2 = {}, acc3 = {};
    for (int k0 = 0; k0 < H; k0 += 32) {
      BFrag bf;
      const float4* kp4 = reinterpret_cast<const float4*>(krow + k0);
#pragma unroll
      for (int j = 0; j < 4; ++j) {
        float4 f = kp4[j];
        bf.b[4 * j + 0] = (__bf16)f.x;
        bf.b[4 * j + 1] = (__bf16)f.y;
        bf.b[4 * j + 2] = (__bf16)f.z;
        bf.b[4 * j + 3] = (__bf16)f.w;
      }
      __builtin_prefetch(krow + k0 + 256, 0, 0);   // global_prefetch_b8
#pragma unroll
      for (int m = 0; m < 4; ++m) {
        const __bf16* ap = qb + (size_t)(16 * m + ln) * H + k0 + 16 * kh;
        AFrag af;
        af.q[0] = *reinterpret_cast<const uint4*>(ap);        // K = c .. c+7
        af.q[1] = *reinterpret_cast<const uint4*>(ap + 8);    // K = c+8 ..
        v8f& acc = (m == 0) ? acc0 : (m == 1) ? acc1 : (m == 2) ? acc2 : acc3;
        acc = __builtin_amdgcn_wmma_f32_16x16x32_bf16(
            false, af.v, false, bf.v, (short)0, acc, false, false);
      }
    }
#if USE_ASYNC_LDS
    __builtin_amdgcn_s_wait_asynccnt(0);   // side-loads done (also pre-barrier)
#endif
    float wm = (1.0f - (float)s_wm[col]) * NEGV;
#pragma unroll
    for (int m = 0; m < 4; ++m) {
      v8f acc = (m == 0) ? acc0 : (m == 1) ? acc1 : (m == 2) ? acc2 : acc3;
#pragma unroll
      for (int j = 0; j < 8; ++j)
        s_scores[16 * m + j + 8 * kh][col] = acc[j] + wm;   // D layout: M=j(+8)
    }
  }
  __syncthreads();

  // ---- Phase 2: masked row softmax -> bf16 P in LDS
  if (threadIdx.x < QS) {
    int r = threadIdx.x;
    float mx = s_scores[r][0];
    for (int c = 1; c < S2; ++c) mx = fmaxf(mx, s_scores[r][c]);
    float sum = 0.0f;
    for (int c = 0; c < S2; ++c) {
      float e = __expf(s_scores[r][c] - mx);
      s_scores[r][c] = e;
      sum += e;
    }
    float inv = 1.0f / sum;
    for (int c = 0; c < S2; ++c)
      s_p[r][c] = (__bf16)(s_scores[r][c] * inv);
  }
  __syncthreads();

  // ---- Phase 3: H = P @ V, scale by sent_prob, atomic accumulate over s
  for (int t = 0; t < 6; ++t) {
    int n0 = 16 * (wave + 8 * t);
    int hc = n0 + ln;                       // this lane's h column (N)
    v8f acc0 = {}, acc1 = {}, acc2 = {}, acc3 = {};
#pragma unroll
    for (int k0 = 0; k0 < S2; k0 += 32) {
      BFrag bf;
      const float* vp = Vb + (size_t)(k0 + 16 * kh) * H + hc;
#pragma unroll
      for (int kk = 0; kk < 16; ++kk)       // 16 lanes x contig f32 per row
        bf.b[kk] = (__bf16)vp[(size_t)kk * H];
#pragma unroll
      for (int m = 0; m < 4; ++m) {
        const __bf16* ap = &s_p[16 * m + ln][k0 + 16 * kh];
        AFrag af;
        af.q[0] = *reinterpret_cast<const uint4*>(ap);
        af.q[1] = *reinterpret_cast<const uint4*>(ap + 8);
        v8f& acc = (m == 0) ? acc0 : (m == 1) ? acc1 : (m == 2) ? acc2 : acc3;
        acc = __builtin_amdgcn_wmma_f32_16x16x32_bf16(
            false, af.v, false, bf.v, (short)0, acc, false, false);
      }
    }
#pragma unroll
    for (int m = 0; m < 4; ++m) {
      v8f acc = (m == 0) ? acc0 : (m == 1) ? acc1 : (m == 2) ? acc2 : acc3;
#pragma unroll
      for (int j = 0; j < 8; ++j) {
        int q = 16 * m + j + 8 * kh;
        atomicAdd(&out[(size_t)(b * QS + q) * H + hc], acc[j] * s_sp[q]);
      }
    }
  }
}

// ----------------------------------------------------------------------------
extern "C" void kernel_launch(void* const* d_in, const int* in_sizes, int n_in,
                              void* d_out, int out_size, void* d_ws, size_t ws_size,
                              hipStream_t stream) {
  const float* query         = (const float*)d_in[0];
  const float* key           = (const float*)d_in[1];
  const float* value         = (const float*)d_in[2];
  const float* key_vec       = (const float*)d_in[3];
  const int*   word_mask     = (const int*)d_in[4];
  const int*   sentence_mask = (const int*)d_in[5];
  float* out = (float*)d_out;

  __bf16* qbf  = (__bf16*)d_ws;
  float* sentp = (float*)((char*)d_ws + (size_t)BS * S1 * H * sizeof(__bf16));

  int n_out = BS * QS * H;
  zero_out_kernel<<<(n_out + 255) / 256, 256, 0, stream>>>(out, n_out);
  prep_kernel<<<BS * QS, 64, 0, stream>>>(query, key_vec, sentence_mask, qbf, sentp);
  attn_kernel<<<BS * S1, 256, 0, stream>>>(qbf, key, value, word_mask, sentp, out);
}
